// FrequencyFirstSystem_20770461844033
// MI455X (gfx1250) — matrix operand order repaired and checked by reading, not compiled, not run on hardware
//
#include <hip/hip_runtime.h>
#include <stdint.h>

typedef _Float16 v16h __attribute__((ext_vector_type(16)));
typedef float    v8f  __attribute__((ext_vector_type(8)));
typedef unsigned int u32x4 __attribute__((ext_vector_type(4)));
typedef int          i32x4 __attribute__((ext_vector_type(4)));
typedef int          i32x8 __attribute__((ext_vector_type(8)));

union F16x16 { v16h v; uint4 q[2]; };

// ---------------- prep: pack weights into WMMA B layout [n][k] f16 ----------
// wB1: [2][64][32]  (k = c*9 + ky*3 + kx, pad 27..31 = 0)
// wB2: [2][128][576] (k = (ky*3+kx)*64 + c)
__global__ void prep_kernel(const float* __restrict__ fw1, const float* __restrict__ tw1,
                            const float* __restrict__ fw2, const float* __restrict__ tw2,
                            _Float16* __restrict__ wB1, _Float16* __restrict__ wB2)
{
    int tid = blockIdx.x * blockDim.x + threadIdx.x;
    int stride = gridDim.x * blockDim.x;
    for (int i = tid; i < 2 * 64 * 32; i += stride) {
        int k = i & 31, n = (i >> 5) & 63, e = i >> 11;
        float v = 0.0f;
        if (k < 27) {
            int c = k / 9, r = k % 9, ky = r / 3, kx = r % 3;
            const float* w = e ? tw1 : fw1;                    // (64,3,3,3)
            v = w[((n * 3 + c) * 3 + ky) * 3 + kx];
        }
        wB1[i] = (_Float16)v;
    }
    for (int i = tid; i < 2 * 128 * 576; i += stride) {
        int k = i % 576, n = (i / 576) & 127, e = i / (128 * 576);
        int tap = k >> 6, c = k & 63, ky = tap / 3, kx = tap % 3;
        const float* w = e ? tw2 : fw2;                        // (128,64,3,3)
        wB2[i] = (_Float16)w[((n * 64 + c) * 3 + ky) * 3 + kx];
    }
}

// ---------------- conv1: 3->64 s2, both experts, implicit GEMM K=32 ---------
// h1 layout: [P*12544][128] f16 NHWC (channel = expert*64 + cout)
__global__ __launch_bounds__(256)
void conv1_kernel(const float* __restrict__ x, const _Float16* __restrict__ wB1,
                  const float* __restrict__ fb1, const float* __restrict__ tb1,
                  _Float16* __restrict__ h1, int img0)
{
    const int lane = threadIdx.x & 31;
    const int wave = threadIdx.x >> 5;
    const int hh   = lane >> 4;           // half-wave select
    const int mrow = lane & 15;
    const int tile = blockIdx.x * 8 + wave;

    // ---- gather A (16x32 im2col rows, f32 -> f16), shared by both experts
    int p   = tile * 16 + mrow;           // output position within chunk
    int bl  = p / 12544, rem = p % 12544;
    int oy  = rem / 112,  ox  = rem % 112;
    const float* xb = x + (size_t)(img0 + bl) * (3 * 224 * 224);

    F16x16 A;
#pragma unroll
    for (int r = 0; r < 16; ++r) {
        int k = hh * 16 + r;                    // {0..15} / {16..31}
        float v = 0.0f;
        if (k < 27) {
            int c = k / 9, t = k % 9, ky = t / 3, kx = t % 3;
            int iy = oy * 2 + ky, ix = ox * 2 + kx;   // SAME: pad_top = 0
            if (iy < 224 && ix < 224) v = xb[(c * 224 + iy) * 224 + ix];
        }
        A.v[r] = (_Float16)v;
    }

#pragma unroll
    for (int e = 0; e < 2; ++e) {
        const float* bias = e ? tb1 : fb1;
#pragma unroll
        for (int nt = 0; nt < 4; ++nt) {
            F16x16 B;
            const uint4* bp = (const uint4*)(wB1 + ((size_t)(e * 64 + nt * 16 + mrow) * 32 + hh * 16));
            B.q[0] = bp[0]; B.q[1] = bp[1];
            v8f acc = {};
            acc = __builtin_amdgcn_wmma_f32_16x16x32_f16(false, A.v, false, B.v,
                                                         (short)0, acc, false, false);
            float bv = bias[nt * 16 + mrow];
#pragma unroll
            for (int r = 0; r < 8; ++r) {
                int m  = hh * 8 + r;                 // D: vgpr r -> M = r / 8+r
                int pp = tile * 16 + m;
                float vv = acc[r] + bv;
                vv = vv > 0.0f ? vv : 0.0f;
                h1[(size_t)pp * 128 + e * 64 + nt * 16 + mrow] = (_Float16)vv;
            }
        }
    }
}

// ---------------- conv2: 64->128 s2 per expert, K=576 (18 WMMA steps) -------
// Weights (32 x 576 f16 slice) staged in LDS via TDM tensor_load_to_lds.
// Deterministic pooled partial sums: partial[e][globalTile][128].
__global__ __launch_bounds__(256)
void conv2_kernel(const _Float16* __restrict__ h1, const _Float16* __restrict__ wB2,
                  const float* __restrict__ fb2, const float* __restrict__ tb2,
                  float* __restrict__ partial, int img0, int P, int nbPerGroup)
{
    __shared__ __align__(16) _Float16 lw[32 * 576];   // 36 KB weight slice

    const int lane = threadIdx.x & 31;
    const int wave = threadIdx.x >> 5;
    const int hh   = lane >> 4;
    const int mrow = lane & 15;
    const int grp  = blockIdx.x / nbPerGroup;    // e*4 + q
    const int e    = grp >> 2, q = grp & 3;
    const int mb   = blockIdx.x % nbPerGroup;

    // ---- TDM: DMA 32*576 f16 = 36864B of wB2[e][q*32..] into LDS (one wave)
    if (wave == 0) {
        uint64_t ga  = (uint64_t)(uintptr_t)(wB2 + (size_t)(e * 128 + q * 32) * 576);
        uint32_t lds = (uint32_t)(uintptr_t)&lw[0];
        u32x4 g0; i32x8 g1; i32x4 g2; i32x4 g3; i32x8 g4;
        g0[0] = 1u;                                   // count=1, user mode
        g0[1] = lds;                                  // lds_addr
        g0[2] = (uint32_t)ga;                         // global_addr[31:0]
        g0[3] = ((uint32_t)(ga >> 32) & 0x01FFFFFFu) | (2u << 30); // addr[56:32] | type=2
        g1[0] = (1 << 16);                            // data_size = 1 (2 bytes)
        g1[1] = (int)(2304u << 16);                   // tensor_dim0 = 2304 (lo16 @63:48)
        g1[2] = (int)(8u << 16);                      // tensor_dim1 = 8    (@95:80)
        g1[3] = (int)(2304u << 16);                   // tile_dim0 = 2304   (@127:112)
        g1[4] = 8;                                    // tile_dim1 = 8      (@143:128)
        g1[5] = 2304;                                 // tensor_dim0_stride (@207:160)
        g1[6] = 0; g1[7] = 0;
        g2[0] = g2[1] = g2[2] = g2[3] = 0;
        g3[0] = g3[1] = g3[2] = g3[3] = 0;
        g4[0] = g4[1] = g4[2] = g4[3] = 0;
        g4[4] = g4[5] = g4[6] = g4[7] = 0;
        __builtin_amdgcn_tensor_load_to_lds(g0, g1, g2, g3, g4, 0);
        __builtin_amdgcn_s_wait_tensorcnt(0);
    }
    __syncthreads();

    const float* bias = e ? tb2 : fb2;
    const float bv0 = bias[q * 32 + mrow];
    const float bv1 = bias[q * 32 + 16 + mrow];

    // hoisted LDS B row bases (per-lane, loop-invariant)
    const _Float16* lw0 = lw + (size_t)mrow        * 576 + hh * 16;
    const _Float16* lw1 = lw + (size_t)(mrow + 16) * 576 + hh * 16;

    const int Mtiles = P * 196;                       // 3136/16 tiles per image
    const int nWaves = nbPerGroup * 8;

    for (int t = mb * 8 + wave; t < Mtiles; t += nWaves) {
        int p  = t * 16 + mrow;
        int bl = p / 3136, rem = p % 3136;
        int oy = rem / 56,  ox  = rem % 56;
        // base pointer folds in expert-channel offset + half-wave K offset
        const _Float16* hb = h1 + (size_t)bl * 12544 * 128 + e * 64 + hh * 16;

        // per-tile tap pointer table (validity folded in as nullptr)
        const _Float16* tp[9];
#pragma unroll
        for (int tap = 0; tap < 9; ++tap) {
            int iy = oy * 2 + tap / 3, ix = ox * 2 + tap % 3;
            tp[tap] = (iy < 112 && ix < 112)
                        ? hb + (size_t)(iy * 112 + ix) * 128
                        : (const _Float16*)0;
        }

        v8f acc0 = {}, acc1 = {};
#pragma unroll
        for (int s = 0; s < 18; ++s) {
            const _Float16* ap0 = tp[s >> 1];
            F16x16 A;
            if (ap0) {
                const uint4* ap = (const uint4*)(ap0 + (s & 1) * 32);
                A.q[0] = ap[0];          // K chunk {+0..7 | +16..23}
                A.q[1] = ap[1];          // K chunk {+8..15 | +24..31}
            } else {
                A.q[0] = make_uint4(0, 0, 0, 0);
                A.q[1] = make_uint4(0, 0, 0, 0);
            }

            F16x16 B0, B1;
            const uint4* b0 = (const uint4*)(lw0 + s * 32);
            const uint4* b1 = (const uint4*)(lw1 + s * 32);
            B0.q[0] = b0[0]; B0.q[1] = b0[1];
            B1.q[0] = b1[0]; B1.q[1] = b1[1];

            acc0 = __builtin_amdgcn_wmma_f32_16x16x32_f16(false, A.v, false, B0.v,
                                                          (short)0, acc0, false, false);
            acc1 = __builtin_amdgcn_wmma_f32_16x16x32_f16(false, A.v, false, B1.v,
                                                          (short)0, acc1, false, false);
        }

        float s0 = 0.0f, s1 = 0.0f;
#pragma unroll
        for (int r = 0; r < 8; ++r) {
            float a = acc0[r] + bv0; s0 += (a > 0.0f ? a : 0.0f);
            float b = acc1[r] + bv1; s1 += (b > 0.0f ? b : 0.0f);
        }
        s0 += __shfl_down(s0, 16, 32);                // combine rows 0-7 with 8-15
        s1 += __shfl_down(s1, 16, 32);
        if (hh == 0) {
            size_t tg = (size_t)img0 * 196 + t;       // global tile index
            partial[((size_t)e * 12544 + tg) * 128 + q * 32 + mrow]      = s0;
            partial[((size_t)e * 12544 + tg) * 128 + q * 32 + 16 + mrow] = s1;
        }
    }
}

// ---------------- reduce: pool means ----------------------------------------
__global__ void reduce_kernel(const float* __restrict__ partial, float* __restrict__ pool)
{
    int i = blockIdx.x * blockDim.x + threadIdx.x;    // over 2*64*128
    if (i >= 2 * 64 * 128) return;
    int n = i & 127, b = (i >> 7) & 63, e = i >> 13;
    const float* pp = partial + ((size_t)e * 12544 + (size_t)b * 196) * 128 + n;
    float s = 0.0f;
    for (int t = 0; t < 196; ++t) s += pp[(size_t)t * 128];
    pool[i] = s * (1.0f / 3136.0f);
}

// ---------------- fc + argmax routing ---------------------------------------
__global__ void fc_route_kernel(const float* __restrict__ pool,
                                const float* __restrict__ fwf, const float* __restrict__ fbf,
                                const float* __restrict__ twf, const float* __restrict__ tbf,
                                float* __restrict__ out)
{
    int b = blockIdx.x * blockDim.x + threadIdx.x;
    if (b >= 64) return;
    float f0 = fbf[0], f1 = fbf[1], t0 = tbf[0], t1 = tbf[1];
    for (int n = 0; n < 128; ++n) {
        float pf = pool[(size_t)b * 128 + n];
        float pt = pool[(size_t)(64 + b) * 128 + n];
        f0 += pf * fwf[n * 2 + 0]; f1 += pf * fwf[n * 2 + 1];
        t0 += pt * twf[n * 2 + 0]; t1 += pt * twf[n * 2 + 1];
    }
    bool real = (f0 >= f1);                            // argmax==0 (first-max ties)
    out[b * 2 + 0] = real ?  100.0f : t0;
    out[b * 2 + 1] = real ? -100.0f : t1;
}

// ---------------- launch ----------------------------------------------------
extern "C" void kernel_launch(void* const* d_in, const int* in_sizes, int n_in,
                              void* d_out, int out_size, void* d_ws, size_t ws_size,
                              hipStream_t stream)
{
    const float* x   = (const float*)d_in[0];
    const float* fw1 = (const float*)d_in[1];
    const float* fb1 = (const float*)d_in[2];
    const float* fw2 = (const float*)d_in[3];
    const float* fb2 = (const float*)d_in[4];
    const float* fwf = (const float*)d_in[5];
    const float* fbf = (const float*)d_in[6];
    const float* tw1 = (const float*)d_in[7];
    const float* tb1 = (const float*)d_in[8];
    const float* tw2 = (const float*)d_in[9];
    const float* tb2 = (const float*)d_in[10];
    const float* twf = (const float*)d_in[11];
    const float* tbf = (const float*)d_in[12];
    float* out = (float*)d_out;

    // workspace layout (bytes)
    const size_t off_wB1     = 0;                     //   8192
    const size_t off_wB2     = 8192;                  // 294912
    const size_t off_partial = 303104;                // 2*12544*128*4 = 12845056
    const size_t off_pool    = 13148160;              // 65536
    const size_t off_h1      = 13213696;              // P*12544*128*2

    char* ws = (char*)d_ws;
    _Float16* wB1   = (_Float16*)(ws + off_wB1);
    _Float16* wB2   = (_Float16*)(ws + off_wB2);
    float* partial  = (float*)(ws + off_partial);
    float* pool     = (float*)(ws + off_pool);
    _Float16* h1    = (_Float16*)(ws + off_h1);

    // chunk batch so h1 fits in workspace (deterministic: depends only on ws_size)
    int P = 64;
    while (P > 1 && off_h1 + (size_t)P * 3211264 > ws_size) P >>= 1;

    prep_kernel<<<64, 256, 0, stream>>>(fw1, tw1, fw2, tw2, wB1, wB2);

    for (int img0 = 0; img0 < 64; img0 += P) {
        conv1_kernel<<<P * 98, 256, 0, stream>>>(x, wB1, fb1, tb1, h1, img0);
        int NB = (P * 196) / 32; if (NB < 1) NB = 1;   // M-blocks per (expert, n-quad)
        conv2_kernel<<<8 * NB, 256, 0, stream>>>(h1, wB2, fb2, tb2, partial, img0, P, NB);
    }

    reduce_kernel<<<(2 * 64 * 128 + 255) / 256, 256, 0, stream>>>(partial, pool);
    fc_route_kernel<<<1, 64, 0, stream>>>(pool, fwf, fbf, twf, tbf, out);
}
